// SwinTransformerBlock_77257871721043
// MI455X (gfx1250) — matrix operand order repaired and checked by reading, not compiled, not run on hardware
//
#include <hip/hip_runtime.h>

// ---------------------------------------------------------------------------
// Swin Transformer block for MI455X (gfx1250, wave32, WMMA).
// All GEMMs: v_wmma_f32_16x16x32_f16. Every WMMA operand is stored in memory
// pre-swizzled into its per-lane register layout, so each fragment load is a
// single contiguous 32-byte read (2 x b128) from global or LDS.
// ---------------------------------------------------------------------------

typedef __attribute__((ext_vector_type(16))) _Float16 v16h;
typedef __attribute__((ext_vector_type(8)))  _Float16 v8h;
typedef __attribute__((ext_vector_type(8)))  float    v8f;

#define WMMA_F16(a, b, c) \
  __builtin_amdgcn_wmma_f32_16x16x32_f16(false, (a), false, (b), (short)0, (c), false, false)

namespace cfg {
constexpr int SS = 3;
constexpr int NWIN = 4096;                 // 64 images * 8*8 windows
constexpr size_t TOKENS = 200704;          // 64*56*56
constexpr size_t TILE   = 512;             // one swizzled 16x32(x16) tile = 32 lanes * 16 halves
constexpr size_t QKVPAN = 2048;            // per (window,head): 4 tiles (64x32)
constexpr size_t APAN   = 8192;            // per 64-row x 128-col A panel: 16 tiles
constexpr size_t HPAN   = 32768;           // per 64-row x 512-col A panel: 64 tiles
}

// ---- swizzle helpers (ISA 7.12.2 wave32 layouts) ---------------------------
// A fragment (16x32 f16): lane = h*16 + (m&15); element e <-> k = 16*(e>>3)+8*h+(e&7)
__device__ __forceinline__ int aswz(int m, int c, int KT) {
  int tile = (m >> 4) * KT + (c >> 5);
  int lane = (((c >> 3) & 1) << 4) | (m & 15);
  int e    = (((c >> 4) & 1) << 3) | (c & 7);
  return (tile * 32 + lane) * 16 + e;
}
// B fragment (32x16 f16): lane = h*16 + n; element e <-> k = 16*(e>>3)+8*h+(e&7)
__device__ __forceinline__ int bswz(int o, int k, int NT) {
  int tile = (k >> 5) * NT + (o >> 4);
  int lane = (((k >> 3) & 1) << 4) | (o & 15);
  return (tile * 32 + lane) * 16 + ((((k >> 4) & 1) << 3) | (k & 7));
}
// contiguous per-lane fragment load: 32 bytes @ lane*32
__device__ __forceinline__ v16h ld_frag(const _Float16* p) {
  return *(const v16h*)(p + (threadIdx.x & 31) * 16);
}

// ---------------------------------------------------------------------------
// Kernel 1: weights -> f16, B-fragment swizzled tile order.
// ---------------------------------------------------------------------------
__global__ void wt_prep_kernel(const float* __restrict__ qkv_w,
                               const float* __restrict__ proj_w,
                               const float* __restrict__ fc1_w,
                               const float* __restrict__ fc2_w,
                               _Float16* __restrict__ qkvT,
                               _Float16* __restrict__ projT,
                               _Float16* __restrict__ fc1T,
                               _Float16* __restrict__ fc2T) {
  int i = blockIdx.x * blockDim.x + threadIdx.x;
  if (i < 384 * 128) qkvT[bswz(i / 128, i % 128, 24)] = (_Float16)qkv_w[i];
  if (i < 128 * 128) projT[bswz(i / 128, i % 128, 8)]  = (_Float16)proj_w[i];
  if (i < 512 * 128) fc1T[bswz(i / 128, i % 128, 32)]  = (_Float16)fc1_w[i];
  if (i < 128 * 512) fc2T[bswz(i / 512, i % 512, 8)]   = (_Float16)fc2_w[i];
}

// ---------------------------------------------------------------------------
// Kernel 2: LN1 + shift(-3,-3) + window partition -> A-swizzled panels
// (NWIN panels of 64x128; rows 49..63 left as-is: padded outputs are unused).
// ---------------------------------------------------------------------------
__global__ __launch_bounds__(128)
void ln1_shift_window_kernel(const float* __restrict__ x,
                             const float* __restrict__ nw,
                             const float* __restrict__ nb,
                             _Float16* __restrict__ yw) {
  __shared__ float red[128];
  int widx = blockIdx.x / 49;
  int n    = blockIdx.x % 49;
  int bimg = widx >> 6;
  int wr = (widx >> 3) & 7, wc = widx & 7;
  int h  = (wr * 7 + n / 7 + cfg::SS) % 56;
  int w  = (wc * 7 + n % 7 + cfg::SS) % 56;
  const float* row = x + (((size_t)bimg * 56 + h) * 56 + w) * 128;
  int t = threadIdx.x;
  float v = row[t];
  red[t] = v; __syncthreads();
#pragma unroll
  for (int s = 64; s > 0; s >>= 1) { if (t < s) red[t] += red[t + s]; __syncthreads(); }
  float mean = red[0] * (1.0f / 128.0f);
  __syncthreads();
  float d = v - mean;
  red[t] = d * d; __syncthreads();
#pragma unroll
  for (int s = 64; s > 0; s >>= 1) { if (t < s) red[t] += red[t + s]; __syncthreads(); }
  float var = red[0] * (1.0f / 128.0f);
  float y = d * rsqrtf(var + 1e-5f) * nw[t] + nb[t];
  yw[(size_t)widx * cfg::APAN + aswz(n, t, 4)] = (_Float16)y;
}

// ---------------------------------------------------------------------------
// Kernel 6: LN2 over x1 -> A-swizzled 64-row panels.
// ---------------------------------------------------------------------------
__global__ __launch_bounds__(128)
void ln2_kernel(const float* __restrict__ x1,
                const float* __restrict__ nw,
                const float* __restrict__ nb,
                _Float16* __restrict__ z) {
  __shared__ float red[128];
  size_t tok = blockIdx.x;
  const float* row = x1 + tok * 128;
  int t = threadIdx.x;
  float v = row[t];
  red[t] = v; __syncthreads();
#pragma unroll
  for (int s = 64; s > 0; s >>= 1) { if (t < s) red[t] += red[t + s]; __syncthreads(); }
  float mean = red[0] * (1.0f / 128.0f);
  __syncthreads();
  float d = v - mean;
  red[t] = d * d; __syncthreads();
#pragma unroll
  for (int s = 64; s > 0; s >>= 1) { if (t < s) red[t] += red[t + s]; __syncthreads(); }
  float var = red[0] * (1.0f / 128.0f);
  float y = d * rsqrtf(var + 1e-5f) * nw[t] + nb[t];
  z[(tok >> 6) * cfg::APAN + aswz((int)(tok & 63), t, 4)] = (_Float16)y;
}

// ---------------------------------------------------------------------------
// Kernel 3: QKV GEMM (per window: 64x128 @ 128x384). No LDS.
// Outputs: Q A-swizzled (for Q@K^T), K,V B-swizzled (as B operands).
// All 64 rows written; padded rows carry bias-only garbage that is never used.
// ---------------------------------------------------------------------------
__global__ __launch_bounds__(256)
void qkv_kernel(const _Float16* __restrict__ yw,
                const _Float16* __restrict__ qkvT,
                const float* __restrict__ qkv_b,
                _Float16* __restrict__ q, _Float16* __restrict__ k,
                _Float16* __restrict__ v) {
  using namespace cfg;
  int widx = blockIdx.x;
  int t = threadIdx.x, wid = t >> 5, lane = t & 31;
  int nn = lane & 15, hh = lane >> 4;
  const _Float16* A = yw + (size_t)widx * APAN;
  for (int tt = wid * 12; tt < wid * 12 + 12; ++tt) {
    int mt = tt / 24, nto = tt % 24;
    v8f acc = {};
#pragma unroll
    for (int ks = 0; ks < 4; ++ks) {
      v16h a  = ld_frag(A + (mt * 4 + ks) * TILE);
      v16h bf = ld_frag(qkvT + (ks * 24 + nto) * TILE);
      acc = WMMA_F16(a, bf, acc);
    }
    int o0 = nto * 16;
    int s = o0 >> 7, head = (o0 >> 5) & 3, hdhi = (o0 >> 4) & 1;
    size_t hb = ((size_t)widx * 4 + head) * QKVPAN;
    float bias = qkv_b[o0 + nn];
    if (s == 0) {                 // Q: A-swizzle, KT=1 (k = head dim, 32 wide)
      int hd = hdhi * 16 + nn;
      int laneQ = ((hd >> 3) & 1) << 4;
      int eQ = ((hd >> 4) << 3) | (hd & 7);
#pragma unroll
      for (int r = 0; r < 8; ++r)
        q[hb + (mt * 32 + laneQ + r + 8 * hh) * 16 + eQ] = (_Float16)(acc[r] + bias);
    } else if (s == 1) {          // K: B-swizzle; token-tile = mt, k = head dim
#pragma unroll
      for (int r = 0; r < 8; ++r)
        k[hb + (mt * 32 + ((nn >> 3) & 1) * 16 + r + 8 * hh) * 16 + (hdhi << 3) + (nn & 7)] = (_Float16)(acc[r] + bias);
    } else {                      // V: B-swizzle; k = token dim -> 8 contiguous halves
      v8h pack;
#pragma unroll
      for (int r = 0; r < 8; ++r) pack[r] = (_Float16)(acc[r] + bias);
      *(v8h*)(v + hb + (((mt >> 1) * 2 + hdhi) * 32 + hh * 16 + nn) * 16 + 8 * (mt & 1)) = pack;
    }
  }
}

// ---------------------------------------------------------------------------
// Kernel 4: attention per (window, head), 64 thr (2 waves).
// Q/K/V fragments loaded directly from pre-swizzled global buffers.
// ---------------------------------------------------------------------------
__global__ __launch_bounds__(64)
void attn_kernel(const _Float16* __restrict__ qg,
                 const _Float16* __restrict__ kg,
                 const _Float16* __restrict__ vg,
                 const float* __restrict__ rpb,
                 _Float16* __restrict__ ab) {
  using namespace cfg;
  __shared__ float    Sf[64 * 64];     // 16 KB scores
  __shared__ _Float16 Psw[8 * 512];    // 8 KB  softmax probs, A-swizzled (KT=2)
  int widx = blockIdx.x >> 2;
  int head = blockIdx.x & 3;
  int wr = (widx >> 3) & 7, wc = widx & 7;
  size_t hb = ((size_t)widx * 4 + head) * QKVPAN;
  int t = threadIdx.x, wid = t >> 5, lane = t & 31;
  int nn = lane & 15, hh = lane >> 4;
  // ---- S = Q @ K^T : 4m x 4n tiles, one wmma each --------------------------
  for (int tt = wid * 8; tt < wid * 8 + 8; ++tt) {
    int mt = tt >> 2, nt = tt & 3;
    v16h a  = ld_frag(qg + hb + mt * TILE);
    v16h bf = ld_frag(kg + hb + nt * TILE);
    v8f acc = {};
    acc = WMMA_F16(a, bf, acc);
#pragma unroll
    for (int r = 0; r < 8; ++r)
      Sf[(mt * 16 + r + 8 * hh) * 64 + nt * 16 + nn] = acc[r];
  }
  __syncthreads();
  // ---- bias + shifted-window mask + softmax, one row per thread ------------
  if (t < 49) {
    int i = t, ri = i / 7, ci = i % 7;
    int hi = wr * 7 + ri, wi = wc * 7 + ci;
    int regi = (hi < 49 ? 0 : (hi < 53 ? 1 : 2)) * 3 + (wi < 49 ? 0 : (wi < 53 ? 1 : 2));
    float buf[49];
    float mx = -1e30f;
    for (int j = 0; j < 49; ++j) {
      int rj = j / 7, cj = j % 7;
      int hj = wr * 7 + rj, wj = wc * 7 + cj;
      int regj = (hj < 49 ? 0 : (hj < 53 ? 1 : 2)) * 3 + (wj < 49 ? 0 : (wj < 53 ? 1 : 2));
      int ridx = (ri - rj + 6) * 13 + (ci - cj + 6);
      float s = Sf[i * 64 + j] * 5.65685424949238f       // *sqrt(HD), faithful to ref
              + rpb[ridx * 4 + head]
              + (regi == regj ? 0.f : -100.f);
      buf[j] = s;
      mx = fmaxf(mx, s);
    }
    float sum = 0.f;
    for (int j = 0; j < 49; ++j) { float e = __expf(buf[j] - mx); buf[j] = e; sum += e; }
    float inv = 1.f / sum;
    for (int j = 0; j < 49; ++j) Psw[aswz(i, j, 2)] = (_Float16)(buf[j] * inv);
    for (int j = 49; j < 64; ++j) Psw[aswz(i, j, 2)] = (_Float16)0.f;
  } else {
    for (int j = 0; j < 64; ++j) Psw[aswz(t, j, 2)] = (_Float16)0.f;
  }
  __syncthreads();
  // ---- O = P @ V : 4m x 2n tiles, 2 k-steps --------------------------------
  size_t wb = (size_t)widx * APAN;
  for (int tt = wid * 4; tt < wid * 4 + 4; ++tt) {
    int mt = tt >> 1, nt = tt & 1;
    v8f acc = {};
#pragma unroll
    for (int ks = 0; ks < 2; ++ks) {
      v16h a  = ld_frag(&Psw[(mt * 2 + ks) * TILE]);
      v16h bf = ld_frag(vg + hb + (ks * 2 + nt) * TILE);
      acc = WMMA_F16(a, bf, acc);
    }
    int c = head * 32 + nt * 16 + nn;           // output column 0..127
    int ks2 = c >> 5;
    int laneB = ((c >> 3) & 1) << 4;
    int e = (((c >> 4) & 1) << 3) | (c & 7);
#pragma unroll
    for (int r = 0; r < 8; ++r) {
      int m = mt * 16 + r + 8 * hh;
      if (m < 49)
        ab[wb + ((mt * 4 + ks2) * 32 + laneB + (m & 15)) * 16 + e] = (_Float16)acc[r];
    }
  }
}

// ---------------------------------------------------------------------------
// Kernel 5: proj GEMM + window reverse + un-shift + residual -> x1. No LDS.
// ---------------------------------------------------------------------------
__global__ __launch_bounds__(256)
void proj_kernel(const _Float16* __restrict__ ab,
                 const _Float16* __restrict__ projT,
                 const float* __restrict__ proj_b,
                 const float* __restrict__ x,
                 float* __restrict__ x1) {
  using namespace cfg;
  int widx = blockIdx.x;
  int bimg = widx >> 6, wr = (widx >> 3) & 7, wc = widx & 7;
  int t = threadIdx.x, wid = t >> 5, lane = t & 31;
  int nn = lane & 15, hh = lane >> 4;
  size_t wb = (size_t)widx * APAN;
  for (int tt = wid * 4; tt < wid * 4 + 4; ++tt) {
    int mt = tt >> 3, nt = tt & 7;
    v8f acc = {};
#pragma unroll
    for (int ks = 0; ks < 4; ++ks) {
      v16h a  = ld_frag(ab + wb + (mt * 4 + ks) * TILE);
      v16h bf = ld_frag(projT + (ks * 8 + nt) * TILE);
      acc = WMMA_F16(a, bf, acc);
    }
    int o = nt * 16 + nn;
    float bias = proj_b[o];
#pragma unroll
    for (int r = 0; r < 8; ++r) {
      int m = mt * 16 + r + 8 * hh;
      if (m < 49) {
        int h = (wr * 7 + m / 7 + cfg::SS) % 56;
        int w = (wc * 7 + m % 7 + cfg::SS) % 56;
        size_t idx = (((size_t)bimg * 56 + h) * 56 + w) * 128 + o;
        x1[idx] = acc[r] + bias + x[idx];
      }
    }
  }
}

// ---------------------------------------------------------------------------
// Kernel 7: FC1 + exact GELU -> hmid (A-swizzled 64x512 panels). No LDS.
// ---------------------------------------------------------------------------
__global__ __launch_bounds__(256)
void fc1_kernel(const _Float16* __restrict__ z,
                const _Float16* __restrict__ fc1T,
                const float* __restrict__ fc1_b,
                _Float16* __restrict__ hmid) {
  using namespace cfg;
  int rt = blockIdx.x;
  int t = threadIdx.x, wid = t >> 5, lane = t & 31;
  int nn = lane & 15, hh = lane >> 4;
  const _Float16* A = z + (size_t)rt * APAN;
  _Float16* H = hmid + (size_t)rt * HPAN;
  for (int tt = wid * 16; tt < wid * 16 + 16; ++tt) {
    int mt = tt >> 5, nt = tt & 31;      // 4 m-tiles x 32 n-tiles
    v8f acc = {};
#pragma unroll
    for (int ks = 0; ks < 4; ++ks) {
      v16h a  = ld_frag(A + (mt * 4 + ks) * TILE);
      v16h bf = ld_frag(fc1T + (ks * 32 + nt) * TILE);
      acc = WMMA_F16(a, bf, acc);
    }
    int o = nt * 16 + nn;
    float bias = fc1_b[o];
    int ks2 = o >> 5;
    int laneB = ((o >> 3) & 1) << 4;
    int e = (((o >> 4) & 1) << 3) | (o & 7);
#pragma unroll
    for (int r = 0; r < 8; ++r) {
      float u = acc[r] + bias;
      float g = 0.5f * u * (1.f + erff(u * 0.70710678118654752f));   // exact GELU
      H[((mt * 16 + ks2) * 32 + laneB + r + 8 * hh) * 16 + e] = (_Float16)g;
    }
  }
}

// ---------------------------------------------------------------------------
// Kernel 8: FC2 + residual -> out (f32). No LDS.
// ---------------------------------------------------------------------------
__global__ __launch_bounds__(256)
void fc2_kernel(const _Float16* __restrict__ hmid,
                const _Float16* __restrict__ fc2T,
                const float* __restrict__ fc2_b,
                const float* __restrict__ x1,
                float* __restrict__ out) {
  using namespace cfg;
  int rt = blockIdx.x;
  size_t row0 = (size_t)rt * 64;
  int t = threadIdx.x, wid = t >> 5, lane = t & 31;
  int nn = lane & 15, hh = lane >> 4;
  const _Float16* A = hmid + (size_t)rt * HPAN;
  for (int tt = wid * 4; tt < wid * 4 + 4; ++tt) {
    int mt = tt >> 3, nt = tt & 7;
    v8f acc = {};
#pragma unroll
    for (int ks = 0; ks < 16; ++ks) {
      v16h a  = ld_frag(A + (mt * 16 + ks) * TILE);
      v16h bf = ld_frag(fc2T + (ks * 8 + nt) * TILE);
      acc = WMMA_F16(a, bf, acc);
    }
    int o = nt * 16 + nn;
    float bias = fc2_b[o];
#pragma unroll
    for (int r = 0; r < 8; ++r) {
      size_t m = row0 + mt * 16 + r + 8 * hh;
      out[m * 128 + o] = acc[r] + bias + x1[m * 128 + o];
    }
  }
}

// ---------------------------------------------------------------------------
// Host launcher
// ---------------------------------------------------------------------------
extern "C" void kernel_launch(void* const* d_in, const int* in_sizes, int n_in,
                              void* d_out, int out_size, void* d_ws, size_t ws_size,
                              hipStream_t stream) {
  using namespace cfg;
  const float* x      = (const float*)d_in[0];
  const float* rpb    = (const float*)d_in[1];
  const float* qkv_w  = (const float*)d_in[2];
  const float* qkv_b  = (const float*)d_in[3];
  const float* proj_w = (const float*)d_in[4];
  const float* proj_b = (const float*)d_in[5];
  const float* n1w    = (const float*)d_in[6];
  const float* n1b    = (const float*)d_in[7];
  const float* n2w    = (const float*)d_in[8];
  const float* n2b    = (const float*)d_in[9];
  const float* fc1_w  = (const float*)d_in[10];
  const float* fc1_b  = (const float*)d_in[11];
  const float* fc2_w  = (const float*)d_in[12];
  const float* fc2_b  = (const float*)d_in[13];
  float* out = (float*)d_out;

  // workspace layout (halves): yw | q | k | v | ab | x1(f32) | weights
  constexpr size_t SEG = (size_t)NWIN * APAN;         // 33,554,432 halves each
  _Float16* wsp = (_Float16*)d_ws;
  _Float16* yw  = wsp;                 // windows, A-swizzled (also z later)
  _Float16* qb  = yw + SEG;
  _Float16* kb  = qb + SEG;
  _Float16* vb  = kb + SEG;
  _Float16* ab  = vb + SEG;            // attn out, A-swizzled
  _Float16* z   = yw;                  // alias: yw dead after QKV
  _Float16* hmid = qb;                 // alias: q|k|v|ab dead after proj
  float*    x1  = (float*)(wsp + 5 * SEG);
  _Float16* qkvT  = (_Float16*)(x1 + TOKENS * 128);
  _Float16* projT = qkvT + 384 * 128;
  _Float16* fc1T  = projT + 128 * 128;
  _Float16* fc2T  = fc1T + 512 * 128;
  (void)in_sizes; (void)n_in; (void)out_size; (void)ws_size;

  wt_prep_kernel<<<dim3(256), dim3(256), 0, stream>>>(qkv_w, proj_w, fc1_w, fc2_w,
                                                      qkvT, projT, fc1T, fc2T);
  ln1_shift_window_kernel<<<dim3((unsigned)(NWIN * 49)), dim3(128), 0, stream>>>(
      x, n1w, n1b, yw);
  qkv_kernel<<<dim3((unsigned)NWIN), dim3(256), 0, stream>>>(
      yw, qkvT, qkv_b, qb, kb, vb);
  attn_kernel<<<dim3((unsigned)(NWIN * 4)), dim3(64), 0, stream>>>(
      qb, kb, vb, rpb, ab);
  proj_kernel<<<dim3((unsigned)NWIN), dim3(256), 0, stream>>>(
      ab, projT, proj_b, x, x1);
  ln2_kernel<<<dim3((unsigned)TOKENS), dim3(128), 0, stream>>>(x1, n2w, n2b, z);
  fc1_kernel<<<dim3((unsigned)(TOKENS / 64)), dim3(256), 0, stream>>>(
      z, fc1T, fc1_b, hmid);
  fc2_kernel<<<dim3((unsigned)(TOKENS / 64)), dim3(256), 0, stream>>>(
      hmid, fc2T, fc2_b, x1, out);
}
